// SBasicBlock_979252544079
// MI455X (gfx1250) — matrix operand order repaired and checked by reading, not compiled, not run on hardware
//
#include <hip/hip_runtime.h>
#include <hip/hip_bf16.h>
#include <stdint.h>

// ---------------- problem dims ----------------
constexpr int Bb = 8, Tt = 30, Cc = 64, Hh = 44, Ww = 44;
constexpr int NImg   = Bb * Tt;       // 240 images through the convs
constexpr int HW     = Hh * Ww;       // 1936 pixels
constexpr int PlaneN = NImg * HW;     // 464640 elems per channel (BN population)
constexpr int KTot   = Cc * 9;        // 576 (im2col K)

// ---------------- vector types for WMMA ----------------
typedef __attribute__((ext_vector_type(16))) __bf16       v16bf;
typedef __attribute__((ext_vector_type(8)))  float        v8f;
typedef __attribute__((ext_vector_type(4)))  unsigned int u32x4;
typedef __attribute__((ext_vector_type(8)))  unsigned int u32x8;

static __device__ __forceinline__ unsigned short f2bf(float f) {
    unsigned int u = __builtin_bit_cast(unsigned int, f);
    unsigned int r = (u + 0x7FFFu + ((u >> 16) & 1u)) >> 16;   // RNE
    return (unsigned short)r;
}

// ---------------- weights fp32 [co][ci][3][3] -> bf16 [co][r][ci] ----------------
__global__ void conv_w_to_bf16(const float* __restrict__ src,
                               unsigned short* __restrict__ dst) {
    int i = blockIdx.x * 256 + threadIdx.x;           // over 64*9*64
    if (i >= Cc * 9 * Cc) return;
    int co = i / (9 * Cc);
    int rem = i % (9 * Cc);
    int r = rem / Cc, ci = rem % Cc;
    dst[co * KTot + r * Cc + ci] = f2bf(src[(co * Cc + ci) * 9 + r]);
}

// ------- fp32 NCHW -> bf16 NHWC (LDS-tiled transpose, coalesced both sides) -------
__global__ __launch_bounds__(256) void nchw_to_nhwc_bf16(
    const float* __restrict__ src, unsigned short* __restrict__ dst) {
    __shared__ float tile[Ww * 65];                   // +1 pad kills bank conflicts
    const int blk = blockIdx.x;                       // n*Hh + h
    const int n = blk / Hh, h = blk % Hh;
    for (int idx = threadIdx.x; idx < Cc * Ww; idx += 256) {
        int ci = idx / Ww, w = idx % Ww;
        tile[w * 65 + ci] = src[(size_t)(n * Cc + ci) * HW + h * Ww + w];
    }
    __syncthreads();
    for (int idx = threadIdx.x; idx < Ww * Cc; idx += 256) {
        int w = idx >> 6, ci = idx & 63;
        dst[((size_t)(n * HW) + h * Ww + w) * Cc + ci] = f2bf(tile[w * 65 + ci]);
    }
}

// ------------- implicit-GEMM 3x3 conv via v_wmma_f32_16x16x32_bf16 -------------
// xb : bf16 NHWC input  [NImg][HW][C]
// wA : bf16 weights     [C][9][C]   (co, tap, ci)  -> A fragments contiguous
// out: fp32 NCHW output [NImg][C][HW] (pre-BN accumulators)
//
// Register-tiled: each wave computes a 32(co) x 32(pixel) macro-tile as
// 2x2 WMMA tiles, so each A/B fragment feeds 2 WMMAs (4 independent WMMAs
// per K-step hide the fragment-load latency). Block = 8 waves
// (2 co-groups x 4 pixel-groups) -> 64 co x 128 pixels of one image.
__global__ __launch_bounds__(256) void conv3x3_wmma(
    const unsigned short* __restrict__ xb,
    const unsigned short* __restrict__ wA,
    float* __restrict__ out) {
    const int lane = threadIdx.x & 31;
    const int wave = threadIdx.x >> 5;
    const int n    = blockIdx.y;
    const int coBase = (wave & 1) << 5;                   // 0 or 32
    const int pBase  = blockIdx.x * 128 + ((wave >> 1) << 5);
    const int nq   = lane & 15;                           // tile column
    const int p0   = pBase + nq;                          // pixel, tile 0
    const int p1   = p0 + 16;                             // pixel, tile 1
    const bool p0OK = (p0 < HW), p1OK = (p1 < HW);
    const int h0 = p0 / Ww, w0 = p0 % Ww;
    const int h1 = p1 / Ww, w1 = p1 % Ww;
    // A k order used here (contiguous, same as B): lanes<16 hold K 0..15, lanes>=16 hold K 16..31
    const int aSel = (lane < 16) ? 0 : 16;
    // B k order used here (contiguous, same as A): lanes<16 hold K 0..15, lanes>=16 hold K 16..31
    const int bSel = (lane < 16) ? 0 : 16;

    v8f acc00 = {}, acc01 = {}, acc10 = {}, acc11 = {};
#pragma unroll
    for (int r = 0; r < 9; ++r) {
        const int dh = r / 3 - 1, dw = r % 3 - 1;
        const int hi0 = h0 + dh, wi0 = w0 + dw;
        const int hi1 = h1 + dh, wi1 = w1 + dw;
        const bool v0 = p0OK && ((unsigned)hi0 < (unsigned)Hh) && ((unsigned)wi0 < (unsigned)Ww);
        const bool v1 = p1OK && ((unsigned)hi1 < (unsigned)Hh) && ((unsigned)wi1 < (unsigned)Ww);
        const unsigned short* bp0 = xb + (size_t)(n * HW + hi0 * Ww + wi0) * Cc;
        const unsigned short* bp1 = xb + (size_t)(n * HW + hi1 * Ww + wi1) * Cc;
        const unsigned short* ap0 = wA + (coBase + nq) * KTot + r * Cc;       // co tile 0
        const unsigned short* ap1 = wA + (coBase + 16 + nq) * KTot + r * Cc;  // co tile 1
        if (r < 8)  // keep next tap's weights hot in near caches
            __builtin_prefetch(wA + (coBase + nq) * KTot + (r + 1) * Cc, 0, 3);
#pragma unroll
        for (int cb = 0; cb < 2; ++cb) {                  // K = 64 channels = 2 x 32
            const int ciBase = cb * 32;
            union { u32x4 q[2]; u32x8 o; } af0, af1;
            af0.q[0] = *(const u32x4*)(ap0 + ciBase + aSel);
            af0.q[1] = *(const u32x4*)(ap0 + ciBase + 8 + aSel);
            af1.q[0] = *(const u32x4*)(ap1 + ciBase + aSel);
            af1.q[1] = *(const u32x4*)(ap1 + ciBase + 8 + aSel);
            u32x8 br0 = {}, br1 = {};
            if (v0) br0 = *(const u32x8*)(bp0 + ciBase + bSel);
            if (v1) br1 = *(const u32x8*)(bp1 + ciBase + bSel);
            v16bf a0 = __builtin_bit_cast(v16bf, af0.o);
            v16bf a1 = __builtin_bit_cast(v16bf, af1.o);
            v16bf b0 = __builtin_bit_cast(v16bf, br0);
            v16bf b1 = __builtin_bit_cast(v16bf, br1);
            acc00 = __builtin_amdgcn_wmma_f32_16x16x32_bf16(false, a0, false, b0, (short)0, acc00, false, false);
            acc01 = __builtin_amdgcn_wmma_f32_16x16x32_bf16(false, a0, false, b1, (short)0, acc01, false, false);
            acc10 = __builtin_amdgcn_wmma_f32_16x16x32_bf16(false, a1, false, b0, (short)0, acc10, false, false);
            acc11 = __builtin_amdgcn_wmma_f32_16x16x32_bf16(false, a1, false, b1, (short)0, acc11, false, false);
        }
    }
    const int mOff = (lane < 16) ? 0 : 8;   // C/D layout: VGPR j -> M=j (+8 hi lanes)
    if (p0OK) {
#pragma unroll
        for (int j = 0; j < 8; ++j) {
            out[(size_t)(n * Cc + coBase + mOff + j) * HW + p0]      = acc00[j];
            out[(size_t)(n * Cc + coBase + 16 + mOff + j) * HW + p0] = acc10[j];
        }
    }
    if (p1OK) {
#pragma unroll
        for (int j = 0; j < 8; ++j) {
            out[(size_t)(n * Cc + coBase + mOff + j) * HW + p1]      = acc01[j];
            out[(size_t)(n * Cc + coBase + 16 + mOff + j) * HW + p1] = acc11[j];
        }
    }
}

// ---------------- per-channel BN stats (sum, sumsq) ----------------
__global__ __launch_bounds__(256) void bn_stats(const float* __restrict__ buf,
                                                float* __restrict__ partial) {
    __shared__ float s1[256], s2[256];
    const int ch = blockIdx.x;
    const int slice = blockIdx.y;
    const int per = PlaneN / 8;                  // 58080, exact
    float a = 0.f, b = 0.f;
    for (int i = slice * per + threadIdx.x; i < (slice + 1) * per; i += 256) {
        int nimg = i / HW, p = i % HW;
        float v = buf[(size_t)(nimg * Cc + ch) * HW + p];
        a += v; b += v * v;
    }
    s1[threadIdx.x] = a; s2[threadIdx.x] = b;
    __syncthreads();
    for (int st = 128; st > 0; st >>= 1) {
        if (threadIdx.x < st) {
            s1[threadIdx.x] += s1[threadIdx.x + st];
            s2[threadIdx.x] += s2[threadIdx.x + st];
        }
        __syncthreads();
    }
    if (threadIdx.x == 0) {
        atomicAdd(&partial[ch * 2 + 0], s1[0]);
        atomicAdd(&partial[ch * 2 + 1], s2[0]);
    }
}

// fold gamma/beta/mean/var into per-channel scale+bias
__global__ void bn_finalize(const float* __restrict__ partial,
                            const float* __restrict__ g,
                            const float* __restrict__ bcoef,
                            float* __restrict__ sb) {
    int c = threadIdx.x;
    if (c >= Cc) return;
    float inv = 1.0f / (float)PlaneN;
    float mean = partial[2 * c] * inv;
    float var  = partial[2 * c + 1] * inv - mean * mean;
    float sc = g[c] * rsqrtf(var + 1e-5f);
    sb[c]      = sc;
    sb[Cc + c] = bcoef[c] - mean * sc;
}

// ---------------- LIF scan over T (in-place conv buffer -> spikes) ----------------
__global__ __launch_bounds__(256) void lif_scan_k(
    float* __restrict__ cbuf,            // [B*T][C][HW]; rewritten with spikes
    const float* __restrict__ sb,        // scale[64] | bias[64]
    const float* __restrict__ alpha,
    const float* __restrict__ resid) {   // nullptr for layer 1, x for layer 2
    int tid = blockIdx.x * 256 + threadIdx.x;    // over B*C*HW = 991232
    if (tid >= Bb * Cc * HW) return;
    int p  = tid % HW;
    int ch = (tid / HW) % Cc;
    int b  = tid / (HW * Cc);
    float sc = sb[ch], bi = sb[Cc + ch], al = alpha[ch];
    float mem = 0.f, prev = 0.f;
    for (int t = 0; t < Tt; ++t) {
        size_t idx = ((size_t)((b * Tt + t) * Cc + ch)) * HW + p;
        float v = cbuf[idx] * sc + bi;
        if (resid) v += resid[idx];
        mem = mem * al + v - prev;               // VTH = 1
        float s = (mem >= 1.0f) ? 1.0f : 0.0f;
        cbuf[idx] = s;
        prev = s;
    }
}

// ---------------- host orchestration ----------------
extern "C" void kernel_launch(void* const* d_in, const int* in_sizes, int n_in,
                              void* d_out, int out_size, void* d_ws, size_t ws_size,
                              hipStream_t stream) {
    (void)in_sizes; (void)n_in; (void)out_size; (void)ws_size;
    const float* x  = (const float*)d_in[0];
    const float* w1 = (const float*)d_in[1];
    const float* w2 = (const float*)d_in[2];
    const float* g1 = (const float*)d_in[3];
    const float* b1 = (const float*)d_in[4];
    const float* g2 = (const float*)d_in[5];
    const float* b2 = (const float*)d_in[6];
    const float* a1 = (const float*)d_in[7];
    const float* a2 = (const float*)d_in[8];

    const int N = Bb * Tt * Cc * HW;             // 29,736,960 per output
    float* outputs  = (float*)d_out;             // spikes2 (final)
    float* outputs1 = (float*)d_out + N;         // spikes1 (final); holds c1 meanwhile

    // workspace carve-up (256-B aligned bumps)
    char* ws = (char*)d_ws;
    size_t off = 0;
    auto alloc = [&](size_t bytes) -> void* {
        void* pp = ws + off;
        off += (bytes + 255) & ~(size_t)255;
        return pp;
    };
    unsigned short* xb  = (unsigned short*)alloc((size_t)NImg * HW * Cc * 2); // 59.5 MB
    unsigned short* spb = (unsigned short*)alloc((size_t)NImg * HW * Cc * 2); // 59.5 MB
    unsigned short* wA1 = (unsigned short*)alloc((size_t)Cc * 9 * Cc * 2);
    unsigned short* wA2 = (unsigned short*)alloc((size_t)Cc * 9 * Cc * 2);
    float* st  = (float*)alloc(4 * Cc * sizeof(float));  // stats1 | stats2
    float* st1 = st;
    float* st2 = st + 2 * Cc;
    float* sb1 = (float*)alloc(2 * Cc * sizeof(float));
    float* sb2 = (float*)alloc(2 * Cc * sizeof(float));

    hipMemsetAsync(st, 0, 4 * Cc * sizeof(float), stream); // atomics need zeros

    conv_w_to_bf16<<<(Cc * 9 * Cc + 255) / 256, 256, 0, stream>>>(w1, wA1);
    conv_w_to_bf16<<<(Cc * 9 * Cc + 255) / 256, 256, 0, stream>>>(w2, wA2);
    nchw_to_nhwc_bf16<<<NImg * Hh, 256, 0, stream>>>(x, xb);

    dim3 cgrid((HW + 127) / 128, NImg);          // (16, 240) blocks, 8 waves each
    // --- layer 1 ---
    conv3x3_wmma<<<cgrid, 256, 0, stream>>>(xb, wA1, outputs1);
    bn_stats<<<dim3(Cc, 8), 256, 0, stream>>>(outputs1, st1);
    bn_finalize<<<1, 64, 0, stream>>>(st1, g1, b1, sb1);
    lif_scan_k<<<(Bb * Cc * HW) / 256, 256, 0, stream>>>(outputs1, sb1, a1, nullptr);
    // --- layer 2 ---
    nchw_to_nhwc_bf16<<<NImg * Hh, 256, 0, stream>>>(outputs1, spb);
    conv3x3_wmma<<<cgrid, 256, 0, stream>>>(spb, wA2, outputs);
    bn_stats<<<dim3(Cc, 8), 256, 0, stream>>>(outputs, st2);
    bn_finalize<<<1, 64, 0, stream>>>(st2, g2, b2, sb2);
    lif_scan_k<<<(Bb * Cc * HW) / 256, 256, 0, stream>>>(outputs, sb2, a2, x);
}
